// CACE_LR_74929999446497
// MI455X (gfx1250) — matrix operand order repaired and checked by reading, not compiled, not run on hardware
//
#include <hip/hip_runtime.h>
#include <hip/hip_bf16.h>

// ---------------- static config (mirrors reference) ----------------
#define NNODES   10000
#define NEDGES   80000
#define NGRAPHS  16
#define NRBF     6
#define NRADIAL  6
#define NANG     20
#define NCH      9
#define NSYM     6
#define PERNODE  1080          // NRADIAL*NANG*NCH
#define BDIM     324           // NRADIAL*NSYM*NCH
#define FLATD    648           // BDIM*2
#define FPAD     672           // 648 padded to 21*32 for K-blocking
#define CUTOFF   5.5f
#define MP_NORM  0.31622776601683794f   // 1/sqrt(10)
#define RBF_SCALE 0.6030226891555273f   // sqrt(2/5.5)
#define NTILES   625           // NNODES/16
#define W1T_HALFS (64 * FPAD)  // 43008 halves = 86016 bytes

typedef __attribute__((ext_vector_type(16))) _Float16 v16h;
typedef __attribute__((ext_vector_type(8)))  float    v8f;

// monomial tables: (lx,ly,lz) enumeration for l=0..3
__constant__ int c_Lof[NANG] = {0,1,1,1,2,2,2,2,2,2,3,3,3,3,3,3,3,3,3,3};
__constant__ int c_LX [NANG] = {0,1,0,0,2,1,1,0,0,0,3,2,2,1,1,1,0,0,0,0};
__constant__ int c_LY [NANG] = {0,0,1,0,0,1,0,2,1,0,0,1,0,2,1,0,3,2,1,0};
__constant__ int c_LZ [NANG] = {0,0,0,1,0,0,1,0,1,2,0,0,1,0,1,2,0,1,2,3};

__device__ __forceinline__ float ipow3(float x, int e) {
    float r = 1.f;
    if (e >= 1) r = x;
    if (e >= 2) r *= x;
    if (e >= 3) r *= x;
    return r;
}

// B[6] from A(a) = base[a*stride]
__device__ __forceinline__ void symmetrize_col(const float* base, int st, float o[6]) {
    float A[NANG];
#pragma unroll
    for (int a = 0; a < NANG; ++a) A[a] = base[a * st];
    o[0] = A[0];
    o[1] = A[1]*A[1] + A[2]*A[2] + A[3]*A[3];
    o[2] = A[4]*A[4] + 2.f*A[5]*A[5] + 2.f*A[6]*A[6] + A[7]*A[7] + 2.f*A[8]*A[8] + A[9]*A[9];
    o[3] = A[10]*A[10] + 3.f*A[11]*A[11] + 3.f*A[12]*A[12] + 3.f*A[13]*A[13] + 6.f*A[14]*A[14]
         + 3.f*A[15]*A[15] + A[16]*A[16] + 3.f*A[17]*A[17] + 3.f*A[18]*A[18] + A[19]*A[19];
    // nu3 (l1=1,l2=1)
    o[4] = A[1]*A[1]*A[4] + A[2]*A[2]*A[7] + A[3]*A[3]*A[9]
         + 4.f*A[1]*A[2]*A[5] + 4.f*A[1]*A[3]*A[6] + 4.f*A[2]*A[3]*A[8];
    // nu3 (l1=1,l2=2)
    o[5] =      A[1]*A[4]*A[10] + 3.f*A[1]*A[5]*A[11] + 3.f*A[1]*A[6]*A[12]
         + 3.f*A[1]*A[7]*A[13] + 6.f*A[1]*A[8]*A[14] + 3.f*A[1]*A[9]*A[15]
         + 3.f*A[2]*A[4]*A[11] + 3.f*A[2]*A[5]*A[13] + 6.f*A[2]*A[6]*A[14]
         +      A[2]*A[7]*A[16] + 3.f*A[2]*A[8]*A[17] + 3.f*A[2]*A[9]*A[18]
         + 3.f*A[3]*A[4]*A[12] + 6.f*A[3]*A[5]*A[14] + 3.f*A[3]*A[6]*A[15]
         + 3.f*A[3]*A[7]*A[17] + 3.f*A[3]*A[8]*A[18] +      A[3]*A[9]*A[19];
}

__device__ __forceinline__ float siluf(float x) { return x / (1.f + __expf(-x)); }

// ---------------- edge geometry helper (per lane, redundant across wave) ---------
struct EdgeGeo { float r, ux, uy, uz, fc; int is, id; };
__device__ __forceinline__ EdgeGeo edge_geo(int e, const float* pos, const int* srcI,
                                            const int* dstI, const float* shifts) {
    EdgeGeo g;
    g.is = srcI[e]; g.id = dstI[e];
    float vx = pos[g.id*3+0] - pos[g.is*3+0] + shifts[e*3+0];
    float vy = pos[g.id*3+1] - pos[g.is*3+1] + shifts[e*3+1];
    float vz = pos[g.id*3+2] - pos[g.is*3+2] + shifts[e*3+2];
    g.r = sqrtf(vx*vx + vy*vy + vz*vz) + 1e-9f;
    g.ux = vx / g.r; g.uy = vy / g.r; g.uz = vz / g.r;
    float u = g.r * (1.f / CUTOFF);
    g.fc = 0.f;
    if (u < 1.f) {
        float u2 = u*u, u3 = u2*u, u6 = u3*u3;
        g.fc = 1.f - 28.f*u6 + 48.f*u6*u - 21.f*u6*u2;   // P=6 polynomial cutoff
    }
    return g;
}

// ================== kernel 1: first scatter  A0[dst] += radial*ang*code ==========
__global__ void edge_scatter1(const float* __restrict__ pos, const int* __restrict__ node_type,
                              const int* __restrict__ srcI, const int* __restrict__ dstI,
                              const float* __restrict__ shifts, const float* __restrict__ Wemb,
                              const float* __restrict__ freqs, float* __restrict__ accA) {
    __shared__ float srad[8][NRBF];
    __shared__ float sang[8][NANG];
    __shared__ float scode[8][NCH];
    const int wid = threadIdx.x >> 5, lane = threadIdx.x & 31;
    const int e = blockIdx.x * 8 + wid;                 // grid sized so e < NEDGES always
    EdgeGeo g = edge_geo(e, pos, srcI, dstI, shifts);
    if (lane < NRBF)
        srad[wid][lane] = RBF_SCALE * __sinf(g.r * freqs[lane]) / g.r * g.fc;
    if (lane < NANG)
        sang[wid][lane] = ipow3(g.ux, c_LX[lane]) * ipow3(g.uy, c_LY[lane]) * ipow3(g.uz, c_LZ[lane]);
    if (lane < NCH)
        scode[wid][lane] = Wemb[node_type[g.is]*3 + lane/3] * Wemb[node_type[g.id]*3 + lane%3];
    __syncthreads();
    float* dbase = accA + (size_t)g.id * PERNODE;
    for (int t = lane; t < PERNODE; t += 32) {
        int ri = t / 180, rem = t - ri*180, ai = rem / NCH, ci = rem - ai*NCH;
        atomicAdd(&dbase[t], srad[wid][ri] * sang[wid][ai] * scode[wid][ci]);
    }
}

// ============ kernel 2: per-node  A=rtf(A0,W_rt)  mem=rtf(A,W_mem)  B0, chi ======
__global__ void node_stage1(float* __restrict__ accA, const float* __restrict__ W_rt,
                            const float* __restrict__ W_mem, float* __restrict__ memb,
                            float* __restrict__ B0, float* __restrict__ chi,
                            const float* __restrict__ W_chi) {
    __shared__ float sA[PERNODE];
    __shared__ float sB[BDIM];
    const int n = blockIdx.x, t = threadIdx.x;
    if (t < 180) {                                     // (a,c) column, in-place radial_tf
        const int a = t / NCH, l = c_Lof[a];
        float in0[NRADIAL], o[NRADIAL];
        float* base = accA + (size_t)n * PERNODE + t;
#pragma unroll
        for (int r = 0; r < NRADIAL; ++r) in0[r] = base[r*180];
#pragma unroll
        for (int s = 0; s < NRADIAL; ++s) {
            float acc = 0.f;
#pragma unroll
            for (int r = 0; r < NRADIAL; ++r) acc += in0[r] * W_rt[l*36 + r*6 + s];
            o[s] = acc;
        }
        float* mbase = memb + (size_t)n * PERNODE + t;
#pragma unroll
        for (int s = 0; s < NRADIAL; ++s) {
            float acc = 0.f;
#pragma unroll
            for (int r = 0; r < NRADIAL; ++r) acc += o[r] * W_mem[l*36 + r*6 + s];
            mbase[s*180] = acc;
            base[s*180]  = o[s];
            sA[s*180+t]  = o[s];
        }
    }
    __syncthreads();
    if (t < 54) {                                      // symmetrize -> B0
        int s = t / NCH, c = t - s*NCH;
        float o[6];
        symmetrize_col(&sA[s*180 + c], NCH, o);
#pragma unroll
        for (int k = 0; k < 6; ++k) {
            int f = (s*6 + k)*NCH + c;
            B0[(size_t)n*BDIM + f] = o[k];
            sB[f] = o[k];
        }
    }
    __syncthreads();
    if (t < NCH) {                                     // chi = B0flat @ W_chi
        float acc = 0.f;
        for (int f = 0; f < BDIM; ++f) acc += sB[f] * W_chi[f*NCH + t];
        chi[(size_t)n*NCH + t] = acc;
    }
}

// ====== kernel 3: second scatter — msg_B into acc1 (r-space), msg_Ar into acc2 ====
__global__ void edge_scatter2(const float* __restrict__ pos, const int* __restrict__ node_type,
                              const int* __restrict__ srcI, const int* __restrict__ dstI,
                              const float* __restrict__ shifts, const float* __restrict__ Wemb,
                              const float* __restrict__ freqs, const float* __restrict__ W_Ar,
                              const float* __restrict__ chi, const float* __restrict__ Amat,
                              float* __restrict__ acc1, float* __restrict__ acc2) {
    __shared__ float srad[8][NRBF];
    __shared__ float sang[8][NANG];
    __shared__ float scode[8][NCH];
    __shared__ float schi[8][NCH];
    __shared__ float sAr[8][NANG*NRADIAL];
    const int wid = threadIdx.x >> 5, lane = threadIdx.x & 31;
    const int e = blockIdx.x * 8 + wid;
    EdgeGeo g = edge_geo(e, pos, srcI, dstI, shifts);
    const float* Asrc = Amat + (size_t)g.is * PERNODE;
    // warm the L2->WGP path for the 4.3KB A[src] gather while we do VALU setup
    // (lane i covers line i*128; 32 lanes span 4096B of the 4320B row)
    __builtin_prefetch((const char*)Asrc + lane * 128, 0, 0);
    if (lane < NRBF)
        srad[wid][lane] = RBF_SCALE * __sinf(g.r * freqs[lane]) / g.r * g.fc;
    if (lane < NANG)
        sang[wid][lane] = ipow3(g.ux, c_LX[lane]) * ipow3(g.uy, c_LY[lane]) * ipow3(g.uz, c_LZ[lane]);
    if (lane < NCH) {
        scode[wid][lane] = Wemb[node_type[g.is]*3 + lane/3] * Wemb[node_type[g.id]*3 + lane%3];
        schi[wid][lane]  = chi[(size_t)g.is*NCH + lane];
    }
    __syncthreads();
    for (int t = lane; t < NANG*NRADIAL; t += 32) {    // rad_Ar[a,s] = sum_r radial[r]*W_Ar[l(a),r,s]
        int a = t / NRADIAL, s = t - a*NRADIAL, l = c_Lof[a];
        float acc = 0.f;
#pragma unroll
        for (int r = 0; r < NRADIAL; ++r) acc += srad[wid][r] * W_Ar[l*36 + r*6 + s];
        sAr[wid][t] = acc;
    }
    __syncthreads();
    float* d1 = acc1 + (size_t)g.id * PERNODE;
    float* d2 = acc2 + (size_t)g.id * PERNODE;
    for (int t = lane; t < PERNODE; t += 32) {
        int si = t / 180, rem = t - si*180, ai = rem / NCH, ci = rem - ai*NCH;
        float v1 = srad[wid][si] * sang[wid][ai] * scode[wid][ci] * schi[wid][ci];
        atomicAdd(&d1[t], v1);                                        // msg_B (r index = si)
        float v2 = Asrc[t] * sAr[wid][ai*NRADIAL + si];               // msg_Ar (s index = si)
        atomicAdd(&d2[t], v2);
    }
}

// === kernel 4: A_new = (acc2 + rtf(acc1))*MP_NORM + mem; B1; feats(f16, padded) ==
__global__ void node_stage2(const float* __restrict__ acc1, const float* __restrict__ acc2,
                            const float* __restrict__ memb, const float* __restrict__ B0,
                            const float* __restrict__ W_rt, _Float16* __restrict__ feats) {
    __shared__ float sA[PERNODE];
    const int n = blockIdx.x, t = threadIdx.x;
    if (t < 180) {
        const int a = t / NCH, l = c_Lof[a];
        float in1[NRADIAL];
#pragma unroll
        for (int r = 0; r < NRADIAL; ++r) in1[r] = acc1[(size_t)n*PERNODE + r*180 + t];
#pragma unroll
        for (int s = 0; s < NRADIAL; ++s) {
            float rtf = 0.f;
#pragma unroll
            for (int r = 0; r < NRADIAL; ++r) rtf += in1[r] * W_rt[l*36 + r*6 + s];
            float v = (acc2[(size_t)n*PERNODE + s*180 + t] + rtf) * MP_NORM
                    + memb[(size_t)n*PERNODE + s*180 + t];
            sA[s*180 + t] = v;
        }
    }
    __syncthreads();
    _Float16* frow = feats + (size_t)n * FPAD;
    if (t < 54) {
        int s = t / NCH, c = t - s*NCH;
        float o[6];
        symmetrize_col(&sA[s*180 + c], NCH, o);
#pragma unroll
        for (int k = 0; k < 6; ++k) {
            int f = (s*6 + k)*NCH + c;
            frow[f*2 + 0] = (_Float16)B0[(size_t)n*BDIM + f];   // stack(..., axis=-1) interleave
            frow[f*2 + 1] = (_Float16)o[k];
        }
    } else if (t < 54 + (FPAD - FLATD)) {
        frow[FLATD + (t - 54)] = (_Float16)0.f;                 // K-padding
    }
}

// ============== kernel 5: weight prep — f16 transpose+pad of W1, W2 ===============
__global__ void prep_weights(const float* __restrict__ W1, const float* __restrict__ W2,
                             _Float16* __restrict__ W1t, _Float16* __restrict__ W2t) {
    int i = blockIdx.x * blockDim.x + threadIdx.x;
    if (i < W1T_HALFS) {
        int nn = i / FPAD, k = i - nn*FPAD;
        W1t[i] = (_Float16)(k < FLATD ? W1[k*64 + nn] : 0.f);
    } else {
        int j = i - W1T_HALFS;
        if (j < 32 * 64) {
            int nn = j / 64, k = j - nn*64;
            W2t[j] = (_Float16)W2[k*32 + nn];
        }
    }
}

// ====== kernel 6: WMMA MLP readout + per-graph reduction (1 wave = 16 nodes) ======
__device__ __forceinline__ v16h load_a_frag(const _Float16* p) {
    v16h a;
#pragma unroll
    for (int j = 0; j < 8; ++j) { a[j] = p[j]; a[8 + j] = p[16 + j]; }
    return a;
}
__device__ __forceinline__ v16h load_b_frag(const _Float16* p) {
    v16h b;
#pragma unroll
    for (int j = 0; j < 8; ++j) { b[j] = p[j]; b[8 + j] = p[16 + j]; }
    return b;
}

__global__ void mlp_readout(const _Float16* __restrict__ feats, const _Float16* __restrict__ W1t,
                            const _Float16* __restrict__ W2t, const float* __restrict__ b1,
                            const float* __restrict__ b2, const float* __restrict__ W3,
                            const float* __restrict__ b3, const int* __restrict__ batch_ids,
                            float* __restrict__ out) {
    __shared__ _Float16 w1s[W1T_HALFS];               // 86 KB: whole W1^T, shared by 8 waves
    __shared__ _Float16 sh1[8][16][64];
    __shared__ float    sh2[8][16][32];

    // ---- async-copy W1^T global -> LDS (ASYNCcnt path), all 256 threads ----
    {
        const unsigned ldsbase = (unsigned)(unsigned long long)(void*)&w1s[0];
        const char* gbase = (const char*)W1t;
        for (int k = 0; k < (W1T_HALFS * 2) / (256 * 16); ++k) {   // 21 iters of 256x16B
            int c = k * 256 + threadIdx.x;
            unsigned la = ldsbase + c * 16;
            const char* ga = gbase + c * 16;
            asm volatile("global_load_async_to_lds_b128 %0, %1, off"
                         :: "v"(la), "v"(ga) : "memory");
        }
        asm volatile("s_wait_asynccnt 0" ::: "memory");
    }
    __syncthreads();

    const int wid = threadIdx.x >> 5, lane = threadIdx.x & 31;
    const int tile = blockIdx.x * 8 + wid;
    if (tile >= NTILES) return;                       // per-wave; no barriers below
    const int mrow = lane & 15, hl = lane >> 4;

    // ---- layer 1: [16,672] x [672,64] via 21 K-blocks of 32, 4 N-tiles ----
    v8f acc[4] = {};
    const _Float16* rowA = feats + (size_t)(tile*16 + mrow) * FPAD;
    for (int kb = 0; kb < FPAD/32; ++kb) {
        v16h a = load_a_frag(rowA + kb*32 + hl*8);
#pragma unroll
        for (int nt = 0; nt < 4; ++nt) {
            v16h b = load_b_frag(&w1s[(nt*16 + mrow) * FPAD + kb*32 + hl*8]);
            acc[nt] = __builtin_amdgcn_wmma_f32_16x16x32_f16(
                false, a, false, b, (short)0, acc[nt], false, false);
        }
    }
#pragma unroll
    for (int nt = 0; nt < 4; ++nt) {
        int col = nt*16 + mrow;
        float bias = b1[col];
#pragma unroll
        for (int i = 0; i < 8; ++i) {
            int m = hl*8 + i;
            sh1[wid][m][col] = (_Float16)siluf(acc[nt][i] + bias);
        }
    }

    // ---- layer 2: [16,64] x [64,32] via 2 K-blocks, 2 N-tiles ----
    v8f acc2[2] = {};
    for (int kb = 0; kb < 2; ++kb) {
        v16h a = load_a_frag(&sh1[wid][mrow][kb*32 + hl*8]);
#pragma unroll
        for (int nt = 0; nt < 2; ++nt) {
            v16h b = load_b_frag(W2t + (size_t)(nt*16 + mrow) * 64 + kb*32 + hl*8);
            acc2[nt] = __builtin_amdgcn_wmma_f32_16x16x32_f16(
                false, a, false, b, (short)0, acc2[nt], false, false);
        }
    }
#pragma unroll
    for (int nt = 0; nt < 2; ++nt) {
        int col = nt*16 + mrow;
        float bias = b2[col];
#pragma unroll
        for (int i = 0; i < 8; ++i)
            sh2[wid][hl*8 + i][col] = siluf(acc2[nt][i] + bias);
    }

    // ---- layer 3 + per-graph segment sum ----
    if (lane < 16) {
        int node = tile*16 + lane;
        float v = b3[0];
#pragma unroll
        for (int c = 0; c < 32; ++c) v += sh2[wid][lane][c] * W3[c];
        atomicAdd(&out[batch_ids[node]], v);
    }
}

// ------------------------------ workspace layout ---------------------------------
static constexpr size_t AL(size_t x) { return (x + 255) & ~(size_t)255; }
static constexpr size_t SZ_ACC  = (size_t)NNODES * PERNODE * sizeof(float);   // 43.2 MB
static constexpr size_t OFF_ACCA  = 0;
static constexpr size_t OFF_MEM   = AL(OFF_ACCA  + SZ_ACC);
static constexpr size_t OFF_ACC1  = AL(OFF_MEM   + SZ_ACC);
static constexpr size_t OFF_ACC2  = AL(OFF_ACC1  + SZ_ACC);
static constexpr size_t OFF_B0    = AL(OFF_ACC2  + SZ_ACC);
static constexpr size_t OFF_CHI   = AL(OFF_B0    + (size_t)NNODES * BDIM * sizeof(float));
static constexpr size_t OFF_FEATS = AL(OFF_CHI   + (size_t)NNODES * NCH * sizeof(float));
static constexpr size_t OFF_W1T   = AL(OFF_FEATS + (size_t)NNODES * FPAD * sizeof(_Float16));
static constexpr size_t OFF_W2T   = AL(OFF_W1T   + (size_t)W1T_HALFS * sizeof(_Float16));
// total ~200 MB

extern "C" void kernel_launch(void* const* d_in, const int* in_sizes, int n_in,
                              void* d_out, int out_size, void* d_ws, size_t ws_size,
                              hipStream_t stream) {
    const float* pos       = (const float*)d_in[0];
    const int*   node_type = (const int*)  d_in[1];
    const int*   srcI      = (const int*)  d_in[2];
    const int*   dstI      = (const int*)  d_in[3];
    const float* shifts    = (const float*)d_in[4];
    const int*   batch_ids = (const int*)  d_in[5];
    const float* Wemb      = (const float*)d_in[6];
    const float* freqs     = (const float*)d_in[7];
    const float* W_rt      = (const float*)d_in[8];
    const float* W_mem     = (const float*)d_in[9];
    const float* W_Ar      = (const float*)d_in[10];
    const float* W_chi     = (const float*)d_in[11];
    const float* W1        = (const float*)d_in[12];
    const float* b1        = (const float*)d_in[13];
    const float* W2        = (const float*)d_in[14];
    const float* b2        = (const float*)d_in[15];
    const float* W3        = (const float*)d_in[16];
    const float* b3        = (const float*)d_in[17];
    float*       out       = (float*)d_out;

    char* ws = (char*)d_ws;
    float*     accA  = (float*)(ws + OFF_ACCA);
    float*     memb  = (float*)(ws + OFF_MEM);
    float*     acc1  = (float*)(ws + OFF_ACC1);
    float*     acc2  = (float*)(ws + OFF_ACC2);
    float*     B0    = (float*)(ws + OFF_B0);
    float*     chi   = (float*)(ws + OFF_CHI);
    _Float16*  feats = (_Float16*)(ws + OFF_FEATS);
    _Float16*  W1t   = (_Float16*)(ws + OFF_W1T);
    _Float16*  W2t   = (_Float16*)(ws + OFF_W2T);

    hipMemsetAsync(accA, 0, SZ_ACC, stream);
    hipMemsetAsync(acc1, 0, SZ_ACC, stream);
    hipMemsetAsync(acc2, 0, SZ_ACC, stream);
    hipMemsetAsync(out,  0, NGRAPHS * sizeof(float), stream);

    prep_weights<<<(W1T_HALFS + 32*64 + 255)/256, 256, 0, stream>>>(W1, W2, W1t, W2t);

    edge_scatter1<<<NEDGES/8, 256, 0, stream>>>(pos, node_type, srcI, dstI, shifts,
                                                Wemb, freqs, accA);
    node_stage1<<<NNODES, 192, 0, stream>>>(accA, W_rt, W_mem, memb, B0, chi, W_chi);
    edge_scatter2<<<NEDGES/8, 256, 0, stream>>>(pos, node_type, srcI, dstI, shifts,
                                                Wemb, freqs, W_Ar, chi, accA, acc1, acc2);
    node_stage2<<<NNODES, 192, 0, stream>>>(acc1, acc2, memb, B0, W_rt, feats);

    mlp_readout<<<(NTILES + 7)/8, 256, 0, stream>>>(feats, W1t, W2t, b1, b2, W3, b3,
                                                    batch_ids, out);
}
